// VitAttention_64003602645322
// MI455X (gfx1250) — matrix operand order repaired and checked by reading, not compiled, not run on hardware
//
#include <hip/hip_runtime.h>
#include <hip/hip_bf16.h>

typedef _Float16 h16 __attribute__((ext_vector_type(16)));
typedef _Float16 h8  __attribute__((ext_vector_type(8)));
typedef float    f8  __attribute__((ext_vector_type(8)));

#define Bsz 64
#define Nn  197
#define Dd  768
#define Hh  12
#define HS  64
#define NP  208   // 13*16, padded sequence
#define NP2 224   // 7*32, padded for PV K-chunks
#define BN  12608 // 64*197, divisible by 16
#define MT  788   // BN/16

__device__ __forceinline__ f8 wmma16(h16 a, h16 b, f8 c) {
  return __builtin_amdgcn_wmma_f32_16x16x32_f16(false, a, false, b, (short)0, c, false, false);
}
__device__ __forceinline__ h16 cat8(h8 lo, h8 hi) {
  return __builtin_shufflevector(lo, hi, 0,1,2,3,4,5,6,7,8,9,10,11,12,13,14,15);
}

// ---------------- weight f32 -> f16 ----------------
__global__ __launch_bounds__(256) void cvt_kernel(const float* __restrict__ src,
                                                  _Float16* __restrict__ dst, int n) {
  int i = (blockIdx.x * 256 + threadIdx.x) * 4;
  if (i < n) {
    float4 f = *(const float4*)(src + i);
    dst[i+0] = (_Float16)f.x; dst[i+1] = (_Float16)f.y;
    dst[i+2] = (_Float16)f.z; dst[i+3] = (_Float16)f.w;
  }
}

// ---------------- LayerNorm (one row per wave) ----------------
__global__ __launch_bounds__(256) void ln_kernel(const float* __restrict__ x,
                                                 const float* __restrict__ g,
                                                 const float* __restrict__ bb,
                                                 _Float16* __restrict__ xn) {
  int row  = blockIdx.x * 8 + (threadIdx.x >> 5);
  int lane = threadIdx.x & 31;
  if (row >= BN) return;
  const float* xr = x + (size_t)row * Dd;
  float v[24]; float s = 0.f, sq = 0.f;
#pragma unroll
  for (int i = 0; i < 24; ++i) { float t = xr[lane + 32*i]; v[i] = t; s += t; sq += t*t; }
#pragma unroll
  for (int off = 16; off; off >>= 1) { s += __shfl_xor(s, off, 32); sq += __shfl_xor(sq, off, 32); }
  float mu  = s  * (1.f/768.f);
  float var = sq * (1.f/768.f) - mu*mu;
  float rs  = rsqrtf(var + 1e-12f);
  _Float16* yr = xn + (size_t)row * Dd;
#pragma unroll
  for (int i = 0; i < 24; ++i) {
    int c = lane + 32*i;
    yr[c] = (_Float16)((v[i] - mu) * rs * g[c] + bb[c]);
  }
}

// ---------------- WMMA GEMM: [BN,768] x [768,768] + bias ----------------
// mode 0: -> Q  [B,H,NP,HS] f16      mode 1: -> K^T [B,H,HS,NP] f16
// mode 2: -> V  [B,H,NP,HS] f16      mode 3: -> out [BN,768] f32
__global__ __launch_bounds__(256) void gemm768(const _Float16* __restrict__ Am,
                                               const _Float16* __restrict__ Wm,
                                               const float* __restrict__ bias,
                                               void* __restrict__ outp, int mode) {
  const int lane  = threadIdx.x & 31;
  const int wave  = threadIdx.x >> 5;
  const int mtile = blockIdx.x * 8 + wave;
  if (mtile >= MT) return;
  const int m0 = mtile * 16;
  const int n0 = blockIdx.y * 64;
  const int lm = lane & 15, lh = lane >> 4;

  f8 acc[4] = {f8{}, f8{}, f8{}, f8{}};
  const _Float16* arow = Am + (size_t)(m0 + lm) * Dd;
  for (int k0 = 0; k0 < Dd; k0 += 32) {
    h8 alo = *(const h8*)(arow + k0 + 8*lh);
    h8 ahi = *(const h8*)(arow + k0 + 16 + 8*lh);
    h16 a  = cat8(alo, ahi);
    const _Float16* wrow = Wm + (size_t)k0 * Dd + n0 + lm;
#pragma unroll
    for (int nt = 0; nt < 4; ++nt) {
      h16 b;
#pragma unroll
      for (int s = 0; s < 16; ++s) b[s] = wrow[(size_t)(8*lh + s + (s >= 8 ? 8 : 0)) * Dd + nt*16];
      acc[nt] = wmma16(a, b, acc[nt]);
    }
  }

  if (mode == 3) {
    float* o = (float*)outp;
#pragma unroll
    for (int r = 0; r < 8; ++r) {
      int m = m0 + r + 8*lh;
#pragma unroll
      for (int nt = 0; nt < 4; ++nt) {
        int col = n0 + nt*16 + lm;
        o[(size_t)m * Dd + col] = acc[nt][r] + bias[col];
      }
    }
  } else {
    _Float16* o = (_Float16*)outp;
#pragma unroll
    for (int r = 0; r < 8; ++r) {
      int m  = m0 + r + 8*lh;
      int bi = m / Nn;
      int n  = m - bi * Nn;
#pragma unroll
      for (int nt = 0; nt < 4; ++nt) {
        int col = n0 + nt*16 + lm;
        int hh = col >> 6, d = col & 63;
        float val = acc[nt][r] + bias[col];
        int idx;
        if (mode == 1) idx = ((bi*Hh + hh)*HS + d)*NP + n;       // K^T
        else           idx = ((bi*Hh + hh)*NP + n)*HS + d;       // Q / V
        o[idx] = (_Float16)val;
      }
    }
  }
}

// ---------------- fused attention: scores -> softmax -> P@V ----------------
__global__ __launch_bounds__(256) void attn_kernel(const _Float16* __restrict__ qg,
                                                   const _Float16* __restrict__ kTg,
                                                   const _Float16* __restrict__ vg,
                                                   _Float16* __restrict__ ctx) {
  extern __shared__ char smem[];
  _Float16* kTl = (_Float16*)smem;                              // [64][208]
  _Float16* Vl  = (_Float16*)(smem + HS*NP*2);                  // [224][64]
  _Float16* Pl  = (_Float16*)(smem + HS*NP*2 + NP2*HS*2);       // [8][16][224]

  const int bh   = blockIdx.x;            // b*12 + h
  const int tid  = threadIdx.x;
  const int lane = tid & 31, wave = tid >> 5;
  const int lm   = lane & 15, lh = lane >> 4;

  // stage K^T fully (incl. garbage pad cols -- masked later)
  {
    const uint4* s4 = (const uint4*)(kTg + (size_t)bh * HS * NP);
    uint4* d4 = (uint4*)kTl;
    for (int i = tid; i < HS*NP/8; i += 256) d4[i] = s4[i];
  }
  // stage V rows 0..196 only; zero rows 197..223 (avoid 0*NaN in PV)
  {
    const uint4* s4 = (const uint4*)(vg + (size_t)bh * NP * HS);
    uint4* d4 = (uint4*)Vl;
    for (int i = tid; i < Nn*HS/8; i += 256) d4[i] = s4[i];     // 197*64/8 = 1576
    uint4 z4 = {0u,0u,0u,0u};
    for (int i = tid; i < (NP2 - Nn)*HS/8; i += 256) d4[Nn*HS/8 + i] = z4;
  }
  __syncthreads();

  _Float16* Pw = Pl + wave * 16 * NP2;

  for (int qt = wave; qt < NP/16; qt += 8) {
    const int q0 = qt * 16;
    const _Float16* qrow = qg + ((size_t)bh * NP + q0 + lm) * HS;
    h16 a0 = cat8(*(const h8*)(qrow +  0 + 8*lh), *(const h8*)(qrow + 16 + 8*lh));
    h16 a1 = cat8(*(const h8*)(qrow + 32 + 8*lh), *(const h8*)(qrow + 48 + 8*lh));

    f8 S[13];
#pragma unroll
    for (int kt = 0; kt < 13; ++kt) {
      h16 b0, b1;
#pragma unroll
      for (int s = 0; s < 16; ++s) {
        int kk = 8*lh + s + (s >= 8 ? 8 : 0);
        b0[s] = kTl[(     kk)*NP + kt*16 + lm];
        b1[s] = kTl[(32 + kk)*NP + kt*16 + lm];
      }
      f8 c = f8{};
      c = wmma16(a0, b0, c);
      c = wmma16(a1, b1, c);
      S[kt] = c;
    }

    // masked softmax per row; rows split: element r <-> row r + 8*lh, col = t*16 + lm
#pragma unroll
    for (int r = 0; r < 8; ++r) {
      float mx = -1e30f;
#pragma unroll
      for (int t = 0; t < 13; ++t) {
        float sv = S[t][r] * 0.125f;   // 1/sqrt(HS)
        S[t][r] = sv;
        if (t*16 + lm < Nn) mx = fmaxf(mx, sv);
      }
#pragma unroll
      for (int off = 1; off < 16; off <<= 1) mx = fmaxf(mx, __shfl_xor(mx, off, 16));
      float sum = 0.f;
#pragma unroll
      for (int t = 0; t < 13; ++t) {
        float e = (t*16 + lm < Nn) ? __expf(S[t][r] - mx) : 0.f;
        S[t][r] = e; sum += e;
      }
#pragma unroll
      for (int off = 1; off < 16; off <<= 1) sum += __shfl_xor(sum, off, 16);
      float inv = 1.f / sum;
      int prow = r + 8*lh;
#pragma unroll
      for (int t = 0; t < 13; ++t)
        Pw[prow*NP2 + t*16 + lm] = (_Float16)(S[t][r] * inv);
    }
    // zero pad cols 208..223 of P
    for (int i = lane; i < 16*16; i += 32)
      Pw[(i >> 4)*NP2 + NP + (i & 15)] = (_Float16)0.f;
    asm volatile("s_wait_dscnt 0" ::: "memory");   // same-wave LDS RAW fence

    // ctx = P @ V  (7 K-chunks of 32 keys)
    f8 acc[4] = {f8{}, f8{}, f8{}, f8{}};
#pragma unroll
    for (int c = 0; c < 7; ++c) {
      int j0 = c * 32;
      h16 ap = cat8(*(const h8*)(Pw + lm*NP2 + j0 + 8*lh),
                    *(const h8*)(Pw + lm*NP2 + j0 + 16 + 8*lh));
#pragma unroll
      for (int nt = 0; nt < 4; ++nt) {
        h16 bv;
#pragma unroll
        for (int s = 0; s < 16; ++s) bv[s] = Vl[(j0 + 8*lh + s + (s >= 8 ? 8 : 0))*HS + nt*16 + lm];
        acc[nt] = wmma16(ap, bv, acc[nt]);
      }
    }
    const int bi = bh / Hh, hh = bh % Hh;
#pragma unroll
    for (int r = 0; r < 8; ++r) {
      int mq = q0 + r + 8*lh;
      if (mq < Nn) {
#pragma unroll
        for (int nt = 0; nt < 4; ++nt)
          ctx[((size_t)(bi*Nn + mq))*Dd + hh*HS + nt*16 + lm] = (_Float16)acc[nt][r];
      }
    }
  }
}

extern "C" void kernel_launch(void* const* d_in, const int* in_sizes, int n_in,
                              void* d_out, int out_size, void* d_ws, size_t ws_size,
                              hipStream_t stream) {
  const float* feat = (const float*)d_in[0];
  const float* Wq = (const float*)d_in[1];  const float* bq = (const float*)d_in[2];
  const float* Wk = (const float*)d_in[3];  const float* bk = (const float*)d_in[4];
  const float* Wv = (const float*)d_in[5];  const float* bv = (const float*)d_in[6];
  const float* Wo = (const float*)d_in[7];  const float* bo = (const float*)d_in[8];
  const float* lg = (const float*)d_in[9];  const float* lb = (const float*)d_in[10];

  char* ws = (char*)d_ws;
  size_t off = 0;
  auto alloc = [&](size_t bytes) -> char* {
    char* p = ws + off;
    off += (bytes + 255) & ~(size_t)255;
    return p;
  };
  _Float16* wq16 = (_Float16*)alloc((size_t)Dd*Dd*2);
  _Float16* wk16 = (_Float16*)alloc((size_t)Dd*Dd*2);
  _Float16* wv16 = (_Float16*)alloc((size_t)Dd*Dd*2);
  _Float16* wo16 = (_Float16*)alloc((size_t)Dd*Dd*2);
  _Float16* xn   = (_Float16*)alloc((size_t)BN*Dd*2);
  _Float16* qb   = (_Float16*)alloc((size_t)Bsz*Hh*NP*HS*2);
  _Float16* kTb  = (_Float16*)alloc((size_t)Bsz*Hh*HS*NP*2);
  _Float16* vb   = (_Float16*)alloc((size_t)Bsz*Hh*NP*HS*2);
  _Float16* ctx  = (_Float16*)alloc((size_t)BN*Dd*2);

  const int nW = Dd*Dd;                 // 589824, = 576*1024
  cvt_kernel<<<576, 256, 0, stream>>>(Wq, wq16, nW);
  cvt_kernel<<<576, 256, 0, stream>>>(Wk, wk16, nW);
  cvt_kernel<<<576, 256, 0, stream>>>(Wv, wv16, nW);
  cvt_kernel<<<576, 256, 0, stream>>>(Wo, wo16, nW);

  ln_kernel<<<BN/8, 256, 0, stream>>>(feat, lg, lb, xn);

  dim3 gg((MT + 7) / 8, Dd / 64);       // (99, 12)
  gemm768<<<gg, 256, 0, stream>>>(xn, wq16, bq, (void*)qb,  0);
  gemm768<<<gg, 256, 0, stream>>>(xn, wk16, bk, (void*)kTb, 1);
  gemm768<<<gg, 256, 0, stream>>>(xn, wv16, bv, (void*)vb,  2);

  size_t shbytes = (size_t)HS*NP*2 + (size_t)NP2*HS*2 + (size_t)8*16*NP2*2; // 112640
  hipFuncSetAttribute((const void*)attn_kernel,
                      hipFuncAttributeMaxDynamicSharedMemorySize, (int)shbytes);
  attn_kernel<<<Bsz*Hh, 256, shbytes, stream>>>(qb, kTb, vb, ctx);

  gemm768<<<gg, 256, 0, stream>>>(ctx, wo16, bo, d_out, 3);
}
